// MultiHeadAttention_59846074302916
// MI455X (gfx1250) — matrix operand order repaired and checked by reading, not compiled, not run on hardware
//
#include <hip/hip_runtime.h>
#include <hip/hip_bf16.h>

#define DM    1024
#define NH    16
#define HDIM  64
#define NB    4
#define SEQ   2048
#define ROWS  (NB*SEQ)          // 8192

// ---- WMMA vector types (gfx1250 wave32) ----
typedef __attribute__((ext_vector_type(16))) __bf16 v16bf;
typedef __attribute__((ext_vector_type(8)))  float  v8f;
typedef __attribute__((ext_vector_type(4)))  unsigned int v4u;
typedef __attribute__((ext_vector_type(8)))  int v8i;
typedef __attribute__((ext_vector_type(4)))  int v4i;

union Frag { v16bf v; uint4 u[2]; };      // one bf16 A/B fragment = 8 VGPRs = 2x16B
union Pk4  { __bf16 h[4]; uint2 u; };

__device__ __forceinline__ v8f wmma_bf16(const v16bf& a, const v16bf& b, const v8f& c) {
    // D = A(16x32 bf16) * B(32x16 bf16) + C(16x16 f32)
    return __builtin_amdgcn_wmma_f32_16x16x32_bf16(false, a, false, b, (short)0, c,
                                                   false, false);
}

#if defined(__has_builtin)
#if __has_builtin(__builtin_amdgcn_tensor_load_to_lds)
#define HAVE_TDM 1
#endif
#endif
#ifndef HAVE_TDM
#define HAVE_TDM 0
#endif

__device__ __forceinline__ void wait_tensor0() {
#if defined(__has_builtin) && __has_builtin(__builtin_amdgcn_s_wait_tensorcnt)
    __builtin_amdgcn_s_wait_tensorcnt(0);
#else
    asm volatile("s_wait_tensorcnt 0x0" ::: "memory");
#endif
}

#if HAVE_TDM
// 2-D TDM tile load: data_size=8B units, LDS pad 16B every 128B row
// (pad_interval code 4 = 32 DWORDs, pad_amount code 3 = 4 DWORDs) -> KSTR=72 bf16 rows.
__device__ __forceinline__ void tdm_load_2d(
    unsigned lds_addr, unsigned long long gaddr,
    unsigned tensor_d0, unsigned tensor_d1,
    unsigned tile_d0, unsigned tile_d1,
    unsigned long long stride0)
{
    v4u g0;
    g0[0] = 1u;                                            // count=1, user descriptor
    g0[1] = lds_addr;                                      // LDS byte address
    g0[2] = (unsigned)(gaddr & 0xFFFFFFFFull);             // global_addr[31:0]
    g0[3] = (unsigned)((gaddr >> 32) & 0x1FFFFFFull)       // global_addr[56:32]
          | (2u << 30);                                    // type=2 (image)
    v8i g1;
    g1[0] = (int)((3u << 16)        // data_size = 8B
                | (1u << 20)        // pad_enable
                | (4u << 22)        // pad_interval: 32 DWORDs (=128B)
                | (3u << 25));      // pad_amount: 4 DWORDs (=16B)
    g1[1] = (int)((tensor_d0 & 0xFFFFu) << 16);            // tensor_dim0[15:0]
    g1[2] = (int)((tensor_d0 >> 16) | ((tensor_d1 & 0xFFFFu) << 16));
    g1[3] = (int)((tensor_d1 >> 16) | (tile_d0 << 16));    // tile_dim0
    g1[4] = (int)(tile_d1 & 0xFFFFu);                      // tile_dim1 (tile_dim2=0)
    g1[5] = (int)(stride0 & 0xFFFFFFFFull);                // tensor_dim0_stride
    g1[6] = (int)((stride0 >> 32) & 0xFFFFull);            // (stride1 = 0)
    g1[7] = 0;
    v4i z4 = {0, 0, 0, 0};                                 // groups 2/3 unused (2-D)
    v8i z8 = {0, 0, 0, 0, 0, 0, 0, 0};                     // trailing group (disabled)
    __builtin_amdgcn_tensor_load_to_lds(g0, g1, z4, z4, z8, 0);
}
#endif

// ---------------- GEMM tiling ----------------
#define BM 128
#define BN 128
#define BK 32
#define LDS_STR 40   // 32 + 8 pad bf16 (80B row stride, multiple of 16B)

// ============================================================
// Kernel 1: QKV = X @ W^T + b ; scatter bf16 into
//   Q [B,H,S,hd], K [B,H,S,hd], V transposed [B,H,hd,S]
// ============================================================
__global__ __launch_bounds__(256) void qkv_gemm_kernel(
    const float* __restrict__ X, const float* __restrict__ W,
    const float* __restrict__ Bias,
    unsigned short* __restrict__ Qp, unsigned short* __restrict__ Kp,
    unsigned short* __restrict__ Vp)
{
    __shared__ __bf16 As[BM * LDS_STR];
    __shared__ __bf16 Bs[BN * LDS_STR];

    __bf16* Qo = (__bf16*)Qp; __bf16* Ko = (__bf16*)Kp; __bf16* Vo = (__bf16*)Vp;

    const int m0 = blockIdx.y * BM, n0 = blockIdx.x * BN;
    const int tid = threadIdx.x, lane = tid & 31, wid = tid >> 5;
    const int wm = wid & 3, wn = wid >> 2;      // 4 waves M x 2 waves N
    const int g = lane >> 4, ln = lane & 15;

    v8f acc[2][4];
    #pragma unroll
    for (int mt = 0; mt < 2; ++mt)
        #pragma unroll
        for (int nt = 0; nt < 4; ++nt)
            #pragma unroll
            for (int r = 0; r < 8; ++r) acc[mt][nt][r] = 0.f;

    for (int k0 = 0; k0 < DM; k0 += BK) {
        if (k0 + BK < DM) {  // warm GL2 for next k-step (global_prefetch_b8)
            __builtin_prefetch(&X[(size_t)(m0 + (tid >> 3)) * DM + k0 + BK + (tid & 7) * 4], 0, 3);
            __builtin_prefetch(&W[(size_t)(n0 + (tid >> 3)) * DM + k0 + BK + (tid & 7) * 4], 0, 3);
        }
        // Stage A tile (fp32 -> bf16): 128 rows x 32 cols
        #pragma unroll
        for (int i = 0; i < 4; ++i) {
            int idx = tid + i * 256;            // 0..1023
            int row = idx >> 3, cg = idx & 7;
            float4 f = *(const float4*)&X[(size_t)(m0 + row) * DM + k0 + cg * 4];
            Pk4 p; p.h[0] = (__bf16)f.x; p.h[1] = (__bf16)f.y;
                   p.h[2] = (__bf16)f.z; p.h[3] = (__bf16)f.w;
            *(uint2*)&As[row * LDS_STR + cg * 4] = p.u;
        }
        // Stage B tile: W rows n0..n0+127, cols k0..k0+31  (stored [N][K])
        #pragma unroll
        for (int i = 0; i < 4; ++i) {
            int idx = tid + i * 256;
            int row = idx >> 3, cg = idx & 7;
            float4 f = *(const float4*)&W[(size_t)(n0 + row) * DM + k0 + cg * 4];
            Pk4 p; p.h[0] = (__bf16)f.x; p.h[1] = (__bf16)f.y;
                   p.h[2] = (__bf16)f.z; p.h[3] = (__bf16)f.w;
            *(uint2*)&Bs[row * LDS_STR + cg * 4] = p.u;
        }
        __syncthreads();

        Frag a[2], b[4];
        #pragma unroll
        for (int mt = 0; mt < 2; ++mt) {
            const __bf16* ar = &As[(wm * 32 + mt * 16 + ln) * LDS_STR];
            a[mt].u[0] = *(const uint4*)&ar[g * 8];          // K = g*8 .. g*8+7
            a[mt].u[1] = *(const uint4*)&ar[16 + g * 8];     // K = 16+g*8 ..
        }
        #pragma unroll
        for (int nt = 0; nt < 4; ++nt) {
            const __bf16* br = &Bs[(wn * 64 + nt * 16 + ln) * LDS_STR];
            b[nt].u[0] = *(const uint4*)&br[g * 8];          // K = g*8 .. g*8+7
            b[nt].u[1] = *(const uint4*)&br[16 + g * 8];     // K = 16+g*8 ..
        }
        #pragma unroll
        for (int mt = 0; mt < 2; ++mt)
            #pragma unroll
            for (int nt = 0; nt < 4; ++nt)
                acc[mt][nt] = wmma_bf16(a[mt].v, b[nt].v, acc[mt][nt]);
        __syncthreads();
    }

    // Epilogue: + bias, scatter bf16 (V goes out transposed [B,H,hd,S])
    #pragma unroll
    for (int nt = 0; nt < 4; ++nt) {
        int n_g = n0 + wn * 64 + nt * 16 + ln;
        float bv = Bias[n_g];
        int which = n_g >> 10;                 // 0=Q 1=K 2=V
        int d = n_g & 1023;
        int h = d >> 6, hdi = d & 63;
        #pragma unroll
        for (int mt = 0; mt < 2; ++mt)
            #pragma unroll
            for (int r = 0; r < 8; ++r) {
                int m_g = m0 + wm * 32 + mt * 16 + r + 8 * g;
                int bI = m_g >> 11, s = m_g & 2047;     // /SEQ, %SEQ
                __bf16 val = (__bf16)(acc[mt][nt][r] + bv);
                if (which == 2)
                    Vo[(((size_t)(bI * NH + h)) * HDIM + hdi) * SEQ + s] = val;
                else {
                    __bf16* dst = (which == 0) ? Qo : Ko;
                    dst[(((size_t)(bI * NH + h)) * SEQ + s) * HDIM + hdi] = val;
                }
            }
    }
}

// ============================================================
// Kernel 2: flash attention, TDM double-buffered K/V tiles.
// 1 block = (b,h) x 128 q rows; 8 waves x 16 rows.
// ============================================================
#define BQ  128
#define BKV 64
#define KSTR 72     // 64 + 8 pad (144B rows; matches TDM pad 16B per 128B)

__global__ __launch_bounds__(256) void attn_kernel(
    const unsigned short* __restrict__ Qp, const unsigned short* __restrict__ Kp,
    const unsigned short* __restrict__ Vp, unsigned short* __restrict__ Cp)
{
    __shared__ __bf16 Ks[2][BKV * KSTR];     // K tile [s][hd], double buffered
    __shared__ __bf16 Vt[2][HDIM * KSTR];    // V tile [hd][s], double buffered
    __shared__ __bf16 Ps[8 * 16 * HDIM];     // per-wave P staging 16x64

    const __bf16* Q = (const __bf16*)Qp;
    const __bf16* K = (const __bf16*)Kp;
    const __bf16* V = (const __bf16*)Vp;     // transposed [B,H,hd,S]
    __bf16*       C = (__bf16*)Cp;

    const int bh = blockIdx.y;
    const int q0 = blockIdx.x * BQ;
    const int tid = threadIdx.x, lane = tid & 31, wid = tid >> 5;
    const int g = lane >> 4, ln = lane & 15;
    const size_t base = (size_t)bh * SEQ * HDIM;   // element base for K and V

    // Resident Q A-fragments (16 rows x hd=64 -> 2 frags)
    Frag qa[2];
    {
        const __bf16* qr = &Q[base + (size_t)(q0 + wid * 16 + ln) * HDIM];
        #pragma unroll
        for (int c = 0; c < 2; ++c) {
            qa[c].u[0] = *(const uint4*)&qr[c * 32 + g * 8];
            qa[c].u[1] = *(const uint4*)&qr[c * 32 + 16 + g * 8];
        }
    }

    // ---- tile stager: TDM DMA (wave 0) or cooperative fallback ----
    auto stage = [&](int kv0, int buf) {
#if HAVE_TDM
        if (wid == 0) {
            unsigned ksA = (unsigned)(unsigned long long)(void*)&Ks[buf][0];
            unsigned vtA = (unsigned)(unsigned long long)(void*)&Vt[buf][0];
            // K tile: rows = 64 seq, row = 64 bf16 = 16 x 8B units
            tdm_load_2d(ksA,
                        (unsigned long long)(const void*)&K[base + (size_t)kv0 * HDIM],
                        /*tensor_d0=*/16, /*tensor_d1=*/SEQ,
                        /*tile_d0=*/16, /*tile_d1=*/BKV,
                        /*stride0=*/16);
            // V tile: rows = 64 hd, row window = 64 bf16 at col kv0 (row len SEQ)
            tdm_load_2d(vtA,
                        (unsigned long long)(const void*)&V[base + (size_t)kv0],
                        /*tensor_d0=*/SEQ / 4, /*tensor_d1=*/HDIM,
                        /*tile_d0=*/16, /*tile_d1=*/HDIM,
                        /*stride0=*/SEQ / 4);
        }
#else
        #pragma unroll
        for (int i = 0; i < 2; ++i) {
            int idx = tid + i * 256;           // 0..511
            int row = idx >> 3, cg = idx & 7;
            *(uint4*)&Ks[buf][row * KSTR + cg * 8] =
                *(const uint4*)&K[base + (size_t)(kv0 + row) * HDIM + cg * 8];
            *(uint4*)&Vt[buf][row * KSTR + cg * 8] =
                *(const uint4*)&V[base + (size_t)row * SEQ + kv0 + cg * 8];
        }
#endif
    };

    float m_i[8], l_i[8];
    v8f o[4];
    #pragma unroll
    for (int r = 0; r < 8; ++r) { m_i[r] = -1e30f; l_i[r] = 0.f; }
    #pragma unroll
    for (int nt = 0; nt < 4; ++nt)
        #pragma unroll
        for (int r = 0; r < 8; ++r) o[nt][r] = 0.f;

    stage(0, 0);                                // prologue: tile 0 -> buf 0
    int cur = 0;

    for (int kv0 = 0; kv0 < SEQ; kv0 += BKV) {
#if HAVE_TDM
        if (wid == 0) wait_tensor0();           // current tile DMA complete
#endif
        __syncthreads();                        // tile visible; prev buf free
        stage((kv0 + BKV < SEQ) ? kv0 + BKV : 0, cur ^ 1);  // overlap next DMA

        const __bf16* ks = &Ks[cur][0];
        const __bf16* vt = &Vt[cur][0];

        // Scores: S = Q * K^T (contraction over hd=64 -> 2 WMMA per n-tile)
        v8f sc[4];
        #pragma unroll
        for (int nt = 0; nt < 4; ++nt) {
            const __bf16* kr = &ks[(nt * 16 + ln) * KSTR];
            Frag b0, b1;
            b0.u[0] = *(const uint4*)&kr[g * 8];
            b0.u[1] = *(const uint4*)&kr[16 + g * 8];
            b1.u[0] = *(const uint4*)&kr[32 + g * 8];
            b1.u[1] = *(const uint4*)&kr[32 + 16 + g * 8];
            v8f z;
            #pragma unroll
            for (int r = 0; r < 8; ++r) z[r] = 0.f;
            z = wmma_bf16(qa[0].v, b0.v, z);
            z = wmma_bf16(qa[1].v, b1.v, z);
            sc[nt] = z;
        }
        #pragma unroll
        for (int nt = 0; nt < 4; ++nt)
            #pragma unroll
            for (int r = 0; r < 8; ++r) sc[nt][r] *= 0.125f;   // 1/sqrt(64)

        // Online softmax: row max / rescale / exp / row sum
        float mnew[8], alpha[8];
        #pragma unroll
        for (int r = 0; r < 8; ++r) {
            float v = fmaxf(fmaxf(sc[0][r], sc[1][r]), fmaxf(sc[2][r], sc[3][r]));
            v = fmaxf(v, __shfl_xor(v, 1, 32));
            v = fmaxf(v, __shfl_xor(v, 2, 32));
            v = fmaxf(v, __shfl_xor(v, 4, 32));
            v = fmaxf(v, __shfl_xor(v, 8, 32));
            mnew[r] = fmaxf(m_i[r], v);
            alpha[r] = __expf(m_i[r] - mnew[r]);
            m_i[r] = mnew[r];
        }
        __bf16* pw = &Ps[wid * 16 * HDIM];
        #pragma unroll
        for (int r = 0; r < 8; ++r) {
            float s = 0.f;
            #pragma unroll
            for (int nt = 0; nt < 4; ++nt) {
                float p = __expf(sc[nt][r] - mnew[r]);
                s += p;
                pw[(r + 8 * g) * HDIM + nt * 16 + ln] = (__bf16)p;  // C -> A layout via LDS
            }
            s += __shfl_xor(s, 1, 32);
            s += __shfl_xor(s, 2, 32);
            s += __shfl_xor(s, 4, 32);
            s += __shfl_xor(s, 8, 32);
            l_i[r] = l_i[r] * alpha[r] + s;
        }
        #pragma unroll
        for (int nt = 0; nt < 4; ++nt)
            #pragma unroll
            for (int r = 0; r < 8; ++r) o[nt][r] *= alpha[r];

        // O += P * V  (contraction over 64 kv positions -> 2 WMMA per hd-tile)
        Frag pa[2];
        {
            const __bf16* pr = &Ps[wid * 16 * HDIM + ln * HDIM];
            #pragma unroll
            for (int c = 0; c < 2; ++c) {
                pa[c].u[0] = *(const uint4*)&pr[c * 32 + g * 8];
                pa[c].u[1] = *(const uint4*)&pr[c * 32 + 16 + g * 8];
            }
        }
        #pragma unroll
        for (int nt = 0; nt < 4; ++nt) {
            const __bf16* vr = &vt[(nt * 16 + ln) * KSTR];
            Frag v0, v1;
            v0.u[0] = *(const uint4*)&vr[g * 8];
            v0.u[1] = *(const uint4*)&vr[16 + g * 8];
            v1.u[0] = *(const uint4*)&vr[32 + g * 8];
            v1.u[1] = *(const uint4*)&vr[32 + 16 + g * 8];
            o[nt] = wmma_bf16(pa[0].v, v0.v, o[nt]);
            o[nt] = wmma_bf16(pa[1].v, v1.v, o[nt]);
        }
        cur ^= 1;
    }

    // Normalize and write context bf16 in [B][S][D] (A-matrix of out proj)
    const int bI = bh >> 4, h = bh & 15;
    #pragma unroll
    for (int r = 0; r < 8; ++r) {
        float inv = 1.f / l_i[r];
        int s = q0 + wid * 16 + r + 8 * g;
        size_t ro = ((size_t)(bI * SEQ + s)) * DM + h * HDIM;
        #pragma unroll
        for (int nt = 0; nt < 4; ++nt)
            C[ro + nt * 16 + ln] = (__bf16)(o[nt][r] * inv);
    }
}

// ============================================================
// Kernel 3: Out = Ctx(bf16) @ out_w^T + out_b  -> fp32 d_out
// ============================================================
__global__ __launch_bounds__(256) void out_gemm_kernel(
    const unsigned short* __restrict__ Ap, const float* __restrict__ W,
    const float* __restrict__ Bias, float* __restrict__ Out)
{
    __shared__ __bf16 As[BM * LDS_STR];
    __shared__ __bf16 Bs[BN * LDS_STR];
    const __bf16* A = (const __bf16*)Ap;

    const int m0 = blockIdx.y * BM, n0 = blockIdx.x * BN;
    const int tid = threadIdx.x, lane = tid & 31, wid = tid >> 5;
    const int wm = wid & 3, wn = wid >> 2;
    const int g = lane >> 4, ln = lane & 15;

    v8f acc[2][4];
    #pragma unroll
    for (int mt = 0; mt < 2; ++mt)
        #pragma unroll
        for (int nt = 0; nt < 4; ++nt)
            #pragma unroll
            for (int r = 0; r < 8; ++r) acc[mt][nt][r] = 0.f;

    for (int k0 = 0; k0 < DM; k0 += BK) {
        if (k0 + BK < DM) {
            __builtin_prefetch(&A[(size_t)(m0 + (tid >> 2)) * DM + k0 + BK + (tid & 3) * 8], 0, 3);
            __builtin_prefetch(&W[(size_t)(n0 + (tid >> 3)) * DM + k0 + BK + (tid & 7) * 4], 0, 3);
        }
        #pragma unroll
        for (int i = 0; i < 2; ++i) {            // A already bf16: 512 x 16B
            int idx = tid + i * 256;
            int row = idx >> 2, cg = idx & 3;
            *(uint4*)&As[row * LDS_STR + cg * 8] =
                *(const uint4*)&A[(size_t)(m0 + row) * DM + k0 + cg * 8];
        }
        #pragma unroll
        for (int i = 0; i < 4; ++i) {            // W fp32 -> bf16
            int idx = tid + i * 256;
            int row = idx >> 3, cg = idx & 7;
            float4 f = *(const float4*)&W[(size_t)(n0 + row) * DM + k0 + cg * 4];
            Pk4 p; p.h[0] = (__bf16)f.x; p.h[1] = (__bf16)f.y;
                   p.h[2] = (__bf16)f.z; p.h[3] = (__bf16)f.w;
            *(uint2*)&Bs[row * LDS_STR + cg * 4] = p.u;
        }
        __syncthreads();

        Frag a[2], b[4];
        #pragma unroll
        for (int mt = 0; mt < 2; ++mt) {
            const __bf16* ar = &As[(wm * 32 + mt * 16 + ln) * LDS_STR];
            a[mt].u[0] = *(const uint4*)&ar[g * 8];
            a[mt].u[1] = *(const uint4*)&ar[16 + g * 8];
        }
        #pragma unroll
        for (int nt = 0; nt < 4; ++nt) {
            const __bf16* br = &Bs[(wn * 64 + nt * 16 + ln) * LDS_STR];
            b[nt].u[0] = *(const uint4*)&br[g * 8];
            b[nt].u[1] = *(const uint4*)&br[16 + g * 8];
        }
        #pragma unroll
        for (int mt = 0; mt < 2; ++mt)
            #pragma unroll
            for (int nt = 0; nt < 4; ++nt)
                acc[mt][nt] = wmma_bf16(a[mt].v, b[nt].v, acc[mt][nt]);
        __syncthreads();
    }

    #pragma unroll
    for (int nt = 0; nt < 4; ++nt) {
        int n_g = n0 + wn * 64 + nt * 16 + ln;
        float bv = Bias[n_g];
        #pragma unroll
        for (int mt = 0; mt < 2; ++mt)
            #pragma unroll
            for (int r = 0; r < 8; ++r) {
                int m_g = m0 + wm * 32 + mt * 16 + r + 8 * g;
                Out[(size_t)m_g * DM + n_g] = acc[mt][nt][r] + bv;
            }
    }
}

// ============================================================
extern "C" void kernel_launch(void* const* d_in, const int* in_sizes, int n_in,
                              void* d_out, int out_size, void* d_ws, size_t ws_size,
                              hipStream_t stream)
{
    (void)in_sizes; (void)n_in; (void)out_size; (void)ws_size;
    const float* x     = (const float*)d_in[0];
    const float* qkv_w = (const float*)d_in[1];
    const float* qkv_b = (const float*)d_in[2];
    const float* out_w = (const float*)d_in[3];
    const float* out_b = (const float*)d_in[4];
    float* out = (float*)d_out;

    const size_t n_per = (size_t)NB * NH * SEQ * HDIM;   // 8.39M elems (bf16)
    unsigned short* Qw = (unsigned short*)d_ws;
    unsigned short* Kw = Qw + n_per;
    unsigned short* Vw = Kw + n_per;                     // V transposed [B,H,hd,S]
    unsigned short* Cw = Vw + n_per;                     // context [B,S,D] bf16

    qkv_gemm_kernel<<<dim3(3 * DM / BN, ROWS / BM), 256, 0, stream>>>(
        x, qkv_w, qkv_b, Qw, Kw, Vw);
    attn_kernel<<<dim3(SEQ / BQ, NB * NH), 256, 0, stream>>>(Qw, Kw, Vw, Cw);
    out_gemm_kernel<<<dim3(DM / BN, ROWS / BM), 256, 0, stream>>>(
        Cw, out_w, out_b, out);
}
